// LSTM_18013092839711
// MI455X (gfx1250) — matrix operand order repaired and checked by reading, not compiled, not run on hardware
//
#include <hip/hip_runtime.h>

typedef __attribute__((ext_vector_type(16))) __bf16  v16bf;
typedef __attribute__((ext_vector_type(8)))  float   v8f;
typedef __attribute__((ext_vector_type(4)))  unsigned int u32x4;
typedef __attribute__((ext_vector_type(4)))  int      v4i;
typedef __attribute__((ext_vector_type(8)))  int      i32x8;
typedef __attribute__((ext_vector_type(4)))  int      i32x4;

#define B_   64
#define T_   512
#define D0_  256
#define H_   512
#define NWG  16
#define NTHREADS 256

union ABPack { u32x4 q[2]; v16bf v; };

#if __has_builtin(__builtin_amdgcn_global_load_async_to_lds_b128)
#define HAVE_ASYNC_LDS 1
#else
#define HAVE_ASYNC_LDS 0
#endif
typedef __attribute__((address_space(1))) v4i gas_v4i;   // global int4
typedef __attribute__((address_space(3))) v4i las_v4i;   // LDS int4

// 16-byte global -> LDS stage; async (DMA-style, ASYNCcnt-tracked) when available.
__device__ __forceinline__ void stage16(unsigned short* ldst, const unsigned short* gsrc) {
#if HAVE_ASYNC_LDS
  __builtin_amdgcn_global_load_async_to_lds_b128((gas_v4i*)gsrc, (las_v4i*)ldst, 0, 0);
#else
  *(u32x4*)ldst = *(const u32x4*)gsrc;
#endif
}

__device__ __forceinline__ void stage_fence() {
#if HAVE_ASYNC_LDS
# if __has_builtin(__builtin_amdgcn_s_wait_asynccnt)
  __builtin_amdgcn_s_wait_asynccnt(0);
# else
  asm volatile("s_wait_asynccnt 0x0" ::: "memory");
# endif
#endif
}

#if __has_builtin(__builtin_amdgcn_tensor_load_to_lds)
#define HAVE_TDM 1
// Tensor Data Mover: DMA a dense tile (rows x rowDwords of 4-byte units, global
// row stride gRowStrideDwords) into contiguous LDS at ldst. rows==1 => 1-D copy.
// D# layout per CDNA5 ISA ch.8: group0 {count=1, lds_addr, global_addr, type=2},
// group1 {data_size=4B, tensor_dim0/1, tile_dim0/1, tensor_dim0_stride}.
__device__ __forceinline__ void tdm_load_2d(const void* gsrc, void* ldst,
                                            unsigned rowDwords, unsigned rows,
                                            unsigned gRowStrideDwords) {
  unsigned long long g = (unsigned long long)gsrc;
  unsigned lds = (unsigned)(unsigned long long)(las_v4i*)ldst;   // LDS byte address
  u32x4 g0;
  g0[0] = 1u;                                                    // count=1, user descriptor
  g0[1] = lds;                                                   // lds_addr[31:0]
  g0[2] = (unsigned)g;                                           // global_addr[31:0]
  g0[3] = (unsigned)((g >> 32) & 0x01FFFFFFu) | 0x80000000u;     // addr[56:32] | type=2
  i32x8 g1;
  g1[0] = (int)(2u << 16);                                       // wg_mask=0, data_size=4B
  g1[1] = (int)((rowDwords & 0xFFFFu) << 16);                    // tensor_dim0[15:0]
  g1[2] = (int)(((rowDwords >> 16) & 0xFFFFu) | ((rows & 0xFFFFu) << 16)); // dim0 hi | tensor_dim1 lo
  g1[3] = (int)(((rows >> 16) & 0xFFFFu) | ((rowDwords & 0xFFFFu) << 16)); // dim1 hi | tile_dim0
  g1[4] = (int)((rows == 1u) ? 0u : (rows & 0xFFFFu));           // tile_dim1 (0 => 1-D) | tile_dim2=0
  g1[5] = (int)gRowStrideDwords;                                 // tensor_dim0_stride[31:0]
  g1[6] = 0;                                                     // stride hi | tensor_dim1_stride lo
  g1[7] = 0;
  i32x4 z4 = {0, 0, 0, 0};
#if __has_include(<hip/amd_detail/amd_gfx1250_TDM.h>)
  i32x8 z8 = {0, 0, 0, 0, 0, 0, 0, 0};                           // clang-23 therock: 6-arg form
  __builtin_amdgcn_tensor_load_to_lds(g0, g1, z4, z4, z8, 0);
#else
  __builtin_amdgcn_tensor_load_to_lds(g0, g1, z4, z4, 0);        // ROCm 7.2: 5-arg form
#endif
}

__device__ __forceinline__ void tdm_wait() {
# if __has_builtin(__builtin_amdgcn_s_wait_tensorcnt)
  __builtin_amdgcn_s_wait_tensorcnt((short)0);
# else
  asm volatile("s_wait_tensorcnt 0x0" ::: "memory");
# endif
}
#else
#define HAVE_TDM 0
#endif

__device__ __forceinline__ v8f wmma_bf16(v16bf a, v16bf b, v8f c) {
  // D = A(16x32) * B(32x16) + C, f32 accumulate
  return __builtin_amdgcn_wmma_f32_16x16x32_bf16(false, a, false, b, (short)0, c, false, false);
}

// A-matrix 16x32 bf16 operand: lane m -> row m, halves K{0..7,16..23}; lane m+16 -> K{8..15,24..31}
__device__ __forceinline__ v16bf load_A(const unsigned short* base, int ldk, int m0, int k0, int L) {
  const unsigned short* p = base + (size_t)(m0 + (L & 15)) * ldk + k0 + ((L >> 4) << 3);
  ABPack r;
  r.q[0] = *(const u32x4*)(p);
  r.q[1] = *(const u32x4*)(p + 16);
  return r.v;
}

// B-matrix 32x16 bf16 operand from row-major W[n,k]: lane n -> col n, K{0..7,16..23}; lane n+16 -> K{8..15,24..31}
__device__ __forceinline__ v16bf load_B(const unsigned short* base, int ldk, int n0, int k0, int L) {
  const unsigned short* p = base + (size_t)(n0 + (L & 15)) * ldk + k0 + ((L >> 4) << 3);
  ABPack r;
  r.q[0] = *(const u32x4*)(p);
  r.q[1] = *(const u32x4*)(p + 16);
  return r.v;
}

__device__ __forceinline__ float sigm(float x) { return 1.0f / (1.0f + __expf(-x)); }
__device__ __forceinline__ float tanh_(float x) {
  float e = __expf(-2.0f * x);
  return (1.0f - e) / (1.0f + e);
}

__device__ __forceinline__ unsigned short f2bf(float f) {
  unsigned int u = __float_as_uint(f);
  u += 0x7FFFu + ((u >> 16) & 1u);   // round-to-nearest-even
  return (unsigned short)(u >> 16);
}

// Device-wide barrier: HW cluster split-barrier when dispatched as a cluster,
// atomic sense barrier otherwise. One sync per LSTM timestep.
__device__ __forceinline__ void grid_sync(unsigned int* cnt, unsigned int* gen, unsigned int target) {
  __threadfence();
  __syncthreads();
  if (__builtin_amdgcn_cluster_id_x() != 0) {
    // Entire 16-WG grid is one workgroup-cluster: s_barrier_signal/-wait -3.
    if (threadIdx.x < 32) __builtin_amdgcn_s_cluster_barrier();
  } else if (threadIdx.x == 0) {
    unsigned int prev = __hip_atomic_fetch_add(cnt, 1u, __ATOMIC_ACQ_REL, __HIP_MEMORY_SCOPE_AGENT);
    if (prev == NWG - 1) {
      __hip_atomic_store(cnt, 0u, __ATOMIC_RELAXED, __HIP_MEMORY_SCOPE_AGENT);
      __hip_atomic_fetch_add(gen, 1u, __ATOMIC_RELEASE, __HIP_MEMORY_SCOPE_AGENT);
    } else {
      while (__hip_atomic_load(gen, __ATOMIC_ACQUIRE, __HIP_MEMORY_SCOPE_AGENT) < target) {
        __builtin_amdgcn_s_sleep(2);
      }
    }
  }
  __syncthreads();
  __threadfence();
}

__global__ void f32_to_bf16_kernel(const float* __restrict__ src, unsigned short* __restrict__ dst, int n) {
  int i = blockIdx.x * blockDim.x + threadIdx.x;
  if (i < n) dst[i] = f2bf(src[i]);
}

// out[b] = hn1[b,:] . fc_w + fc_b   (tiny: 64x512x1)
__global__ void fc_kernel(const float* __restrict__ hn1, const float* __restrict__ w,
                          const float* __restrict__ bfc, float* __restrict__ out) {
  int b = threadIdx.x;
  if (b < B_) {
    float s = bfc[0];
    #pragma unroll 8
    for (int k = 0; k < H_; ++k) s += hn1[(size_t)b * H_ + k] * w[k];
    out[b] = s;
  }
}

// Persistent fused 2-layer LSTM scan.
// Grid: 16 WGs x 256 threads (8 wave32). WG w owns hidden units [32w, 32w+32).
// Wave v: Mtile = v>>1 (batch rows), Ntile = v&1 (units sub-slab of 16), all 4 gates.
__global__ void __launch_bounds__(NTHREADS, 1)
lstm_persistent(const unsigned short* __restrict__ x_bf,
                unsigned short* __restrict__ out0_bf,
                const unsigned short* __restrict__ Wih0,
                const unsigned short* __restrict__ Whh0,
                const unsigned short* __restrict__ Wih1,
                const unsigned short* __restrict__ Whh1,
                const float* __restrict__ bih0, const float* __restrict__ bhh0,
                const float* __restrict__ bih1, const float* __restrict__ bhh1,
                unsigned short* __restrict__ hA, unsigned short* __restrict__ hB,
                float* __restrict__ hn_out, float* __restrict__ cn_out,
                unsigned int* bar_cnt, unsigned int* bar_gen)
{
  extern __shared__ unsigned short smem[];
  unsigned short* Wslab = smem;               // 128 rows (4 gates x 32 units) x 512 K = 128 KB

  const int tid = threadIdx.x;
  const int wg  = blockIdx.x;
  const int wv  = tid >> 5;
  const int L   = tid & 31;
  const int mt  = wv >> 1;          // batch tile 0..3
  const int nt  = wv & 1;           // unit sub-tile 0..1
  const int u0  = wg * 32;          // first hidden unit owned by this WG
  const int un0 = u0 + nt * 16;     // this wave's 16-unit column base
  const int col = un0 + (L & 15);   // this lane's hidden-unit column (C-layout N=lane%16)

  unsigned int target = 0;

  for (int layer = 0; layer < 2; ++layer) {
    const unsigned short* xsrc = layer ? out0_bf : x_bf;   // [B, T, D] bf16
    const int D = layer ? H_ : D0_;
    const unsigned short* Wih = layer ? Wih1 : Wih0;       // [4H, D] bf16
    const unsigned short* Whh = layer ? Whh1 : Whh0;       // [4H, H] bf16
    const float* bih = layer ? bih1 : bih0;
    const float* bhh = layer ? bhh1 : bhh0;

    unsigned short* abx = smem + 128 * H_;    // x_t tile  [64 x D]   (<= 64 KB)
    unsigned short* abh = abx + B_ * D;       // h_{t-1}   [64 x 512] (64 KB)

    // Stage this WG's W_hh slab into LDS once per layer (async mem->LDS; reused 512 steps).
    for (int i = tid; i < (128 * H_ / 8); i += NTHREADS) {
      int r  = i >> 6;               // slab row 0..127
      int k8 = (i & 63) << 3;        // K offset
      stage16(Wslab + r * H_ + k8,
              Whh + (size_t)((r >> 5) * H_ + u0 + (r & 31)) * H_ + k8);
    }

    // Zero this WG's slice of both h double-buffers (h0 = 0).
    {
      const int per = B_ * H_ / NWG;  // 2048 halves
      for (int i = tid; i < per / 8; i += NTHREADS) {
        u32x4 z = {0u, 0u, 0u, 0u};
        *(u32x4*)(hA + wg * per + i * 8) = z;
        *(u32x4*)(hB + wg * per + i * 8) = z;
      }
    }

    // Per-lane folded gate biases (b_ih + b_hh).
    const float bias_i = bih[0 * H_ + col] + bhh[0 * H_ + col];
    const float bias_f = bih[1 * H_ + col] + bhh[1 * H_ + col];
    const float bias_g = bih[2 * H_ + col] + bhh[2 * H_ + col];
    const float bias_o = bih[3 * H_ + col] + bhh[3 * H_ + col];

    v8f cst = {};   // cell-state tile (fp32) lives in VGPRs across the whole scan

    stage_fence();
    ++target; grid_sync(bar_cnt, bar_gen, target);

    for (int t = 0; t < T_; ++t) {
      const unsigned short* hprev = (t & 1) ? hA : hB;
      unsigned short*       hnew  = (t & 1) ? hB : hA;

#if HAVE_TDM
      // Tensor Data Mover: one descriptor per tile, wave 0 issues, TENSORcnt-tracked.
      if (wv == 0) {
        // x_t: 2-D tile, 64 rows of D/2 dwords, global row stride T*D/2 dwords.
        tdm_load_2d(xsrc + (size_t)t * D, abx, (unsigned)(D >> 1), B_,
                    (unsigned)((size_t)T_ * D >> 1));
        // h_{t-1}: flat 1-D copy, 16384 dwords.
        tdm_load_2d(hprev, abh, (B_ * H_) >> 1, 1u, (B_ * H_) >> 1);
        tdm_wait();
      }
#else
      {
        const int DK8 = D >> 3;
        for (int i = tid; i < B_ * DK8; i += NTHREADS) {
          int b  = i / DK8;
          int k8 = (i % DK8) << 3;
          stage16(abx + b * D + k8, xsrc + ((size_t)b * T_ + t) * D + k8);
        }
        for (int i = tid; i < B_ * (H_ >> 3); i += NTHREADS) {
          int b  = i >> 6;
          int k8 = (i & 63) << 3;
          stage16(abh + b * H_ + k8, hprev + b * H_ + k8);
        }
        stage_fence();
      }
#endif
      __syncthreads();

      v8f ai = {}, af = {}, ag = {}, ao = {};

      // ---- Input-projection part: B streamed from global W_ih (2x-unrolled pipeline) ----
      {
        v16bf a0 = load_A(abx, D, mt * 16, 0, L);
        v16bf p0 = load_B(Wih, D, 0 * H_ + un0, 0, L);
        v16bf p1 = load_B(Wih, D, 1 * H_ + un0, 0, L);
        v16bf p2 = load_B(Wih, D, 2 * H_ + un0, 0, L);
        v16bf p3 = load_B(Wih, D, 3 * H_ + un0, 0, L);
        for (int kt = 0; kt < D; kt += 64) {
          const int k1 = kt + 32;
          __builtin_prefetch(Wih + (size_t)(un0 + (L & 15)) * D + k1 + 32, 0, 3);
          v16bf a1 = load_A(abx, D, mt * 16, k1, L);
          v16bf q0 = load_B(Wih, D, 0 * H_ + un0, k1, L);
          v16bf q1 = load_B(Wih, D, 1 * H_ + un0, k1, L);
          v16bf q2 = load_B(Wih, D, 2 * H_ + un0, k1, L);
          v16bf q3 = load_B(Wih, D, 3 * H_ + un0, k1, L);
          ai = wmma_bf16(a0, p0, ai);
          af = wmma_bf16(a0, p1, af);
          ag = wmma_bf16(a0, p2, ag);
          ao = wmma_bf16(a0, p3, ao);
          const int k2 = (kt + 64 < D) ? kt + 64 : 0;   // dummy re-load on last iter
          a0 = load_A(abx, D, mt * 16, k2, L);
          p0 = load_B(Wih, D, 0 * H_ + un0, k2, L);
          p1 = load_B(Wih, D, 1 * H_ + un0, k2, L);
          p2 = load_B(Wih, D, 2 * H_ + un0, k2, L);
          p3 = load_B(Wih, D, 3 * H_ + un0, k2, L);
          ai = wmma_bf16(a1, q0, ai);
          af = wmma_bf16(a1, q1, af);
          ag = wmma_bf16(a1, q2, ag);
          ao = wmma_bf16(a1, q3, ao);
        }
      }
      // ---- Recurrent part: B from LDS-resident W_hh slab (2x-unrolled pipeline) ----
      {
        const int nb = nt * 16;
        v16bf a0 = load_A(abh, H_, mt * 16, 0, L);
        v16bf p0 = load_B(Wslab, H_, 0 * 32 + nb, 0, L);
        v16bf p1 = load_B(Wslab, H_, 1 * 32 + nb, 0, L);
        v16bf p2 = load_B(Wslab, H_, 2 * 32 + nb, 0, L);
        v16bf p3 = load_B(Wslab, H_, 3 * 32 + nb, 0, L);
        for (int kt = 0; kt < H_; kt += 64) {
          const int k1 = kt + 32;
          v16bf a1 = load_A(abh, H_, mt * 16, k1, L);
          v16bf q0 = load_B(Wslab, H_, 0 * 32 + nb, k1, L);
          v16bf q1 = load_B(Wslab, H_, 1 * 32 + nb, k1, L);
          v16bf q2 = load_B(Wslab, H_, 2 * 32 + nb, k1, L);
          v16bf q3 = load_B(Wslab, H_, 3 * 32 + nb, k1, L);
          ai = wmma_bf16(a0, p0, ai);
          af = wmma_bf16(a0, p1, af);
          ag = wmma_bf16(a0, p2, ag);
          ao = wmma_bf16(a0, p3, ao);
          const int k2 = (kt + 64 < H_) ? kt + 64 : 0;
          a0 = load_A(abh, H_, mt * 16, k2, L);
          p0 = load_B(Wslab, H_, 0 * 32 + nb, k2, L);
          p1 = load_B(Wslab, H_, 1 * 32 + nb, k2, L);
          p2 = load_B(Wslab, H_, 2 * 32 + nb, k2, L);
          p3 = load_B(Wslab, H_, 3 * 32 + nb, k2, L);
          ai = wmma_bf16(a1, q0, ai);
          af = wmma_bf16(a1, q1, af);
          ag = wmma_bf16(a1, q2, ag);
          ao = wmma_bf16(a1, q3, ao);
        }
      }

      // Lane-local LSTM cell update: this wave holds i,f,g,o for its (M,N) tile.
      #pragma unroll
      for (int r = 0; r < 8; ++r) {
        float gi = ai[r] + bias_i;
        float gf = af[r] + bias_f;
        float gg = ag[r] + bias_g;
        float go = ao[r] + bias_o;
        float i_ = sigm(gi), f_ = sigm(gf), c_ = tanh_(gg), o_ = sigm(go);
        float cnew = f_ * cst[r] + i_ * c_;
        cst[r] = cnew;
        float h_ = o_ * tanh_(cnew);
        int row = mt * 16 + r + ((L >> 4) << 3);   // C layout: lanes16-31 hold M=8+r
        unsigned short hb = f2bf(h_);
        hnew[row * H_ + col] = hb;
        if (layer == 0) out0_bf[((size_t)row * T_ + t) * H_ + col] = hb;
        if (t == T_ - 1) {
          hn_out[((size_t)layer * B_ + row) * H_ + col] = h_;
          cn_out[((size_t)layer * B_ + row) * H_ + col] = cnew;
        }
      }

      ++target; grid_sync(bar_cnt, bar_gen, target);
    }
  }
}

extern "C" void kernel_launch(void* const* d_in, const int* in_sizes, int n_in,
                              void* d_out, int out_size, void* d_ws, size_t ws_size,
                              hipStream_t stream) {
  (void)in_sizes; (void)n_in; (void)out_size; (void)ws_size;
  const float* x     = (const float*)d_in[0];
  const float* W_ih0 = (const float*)d_in[1];
  const float* W_hh0 = (const float*)d_in[2];
  const float* b_ih0 = (const float*)d_in[3];
  const float* b_hh0 = (const float*)d_in[4];
  const float* W_ih1 = (const float*)d_in[5];
  const float* W_hh1 = (const float*)d_in[6];
  const float* b_ih1 = (const float*)d_in[7];
  const float* b_hh1 = (const float*)d_in[8];
  const float* fc_w  = (const float*)d_in[9];
  const float* fc_b  = (const float*)d_in[10];

  char* ws = (char*)d_ws;
  size_t off = 0;
  auto alloc = [&](size_t bytes) -> void* {
    void* p = ws + off;
    off = (off + bytes + 255) & ~(size_t)255;
    return p;
  };
  unsigned int*   bar  = (unsigned int*)  alloc(256);
  unsigned short* hA   = (unsigned short*)alloc((size_t)B_ * H_ * 2);
  unsigned short* hB   = (unsigned short*)alloc((size_t)B_ * H_ * 2);
  unsigned short* xbf  = (unsigned short*)alloc((size_t)B_ * T_ * D0_ * 2);
  unsigned short* o0   = (unsigned short*)alloc((size_t)B_ * T_ * H_ * 2);
  unsigned short* wih0 = (unsigned short*)alloc((size_t)4 * H_ * D0_ * 2);
  unsigned short* whh0 = (unsigned short*)alloc((size_t)4 * H_ * H_ * 2);
  unsigned short* wih1 = (unsigned short*)alloc((size_t)4 * H_ * H_ * 2);
  unsigned short* whh1 = (unsigned short*)alloc((size_t)4 * H_ * H_ * 2);

  (void)hipMemsetAsync(bar, 0, 256, stream);   // reset grid barrier every launch (graph-capture safe)

  auto conv = [&](const float* s, unsigned short* d, int n) {
    f32_to_bf16_kernel<<<(n + 255) / 256, 256, 0, stream>>>(s, d, n);
  };
  conv(x,     xbf,  B_ * T_ * D0_);
  conv(W_ih0, wih0, 4 * H_ * D0_);
  conv(W_hh0, whh0, 4 * H_ * H_);
  conv(W_ih1, wih1, 4 * H_ * H_);
  conv(W_hh1, whh1, 4 * H_ * H_);

  float* out = (float*)d_out;            // [64]
  float* hn  = out + B_;                 // [2, 64, 512]
  float* cn  = hn + 2 * B_ * H_;         // [2, 64, 512]

  const int smem_bytes = (128 * H_ + B_ * (H_ + H_)) * 2;   // 256 KB dynamic LDS
  (void)hipFuncSetAttribute((const void*)lstm_persistent,
                            hipFuncAttributeMaxDynamicSharedMemorySize, smem_bytes);
  lstm_persistent<<<NWG, NTHREADS, smem_bytes, stream>>>(
      xbf, o0, wih0, whh0, wih1, whh1,
      b_ih0, b_hh0, b_ih1, b_hh1,
      hA, hB, hn, cn, bar, bar + 1);

  fc_kernel<<<1, 64, 0, stream>>>(hn + (size_t)B_ * H_, fc_w, fc_b, out);
}
